// DGAT_61873298866671
// MI455X (gfx1250) — matrix operand order repaired and checked by reading, not compiled, not run on hardware
//
#include <hip/hip_runtime.h>
#include <hip/hip_bf16.h>
#include <cstdint>
#include <cstddef>

// ---------------- problem constants ----------------
#define N_NODES   20000
#define IN_F      32
#define N_EDGES   200000
#define EDGE_INF  16
#define HID       64
#define HEADS     4
#define OUTF      8
#define HO        32
#define NEG_SLOPE 0.2f

#define KSTEPS 65      // 64 hidden units + 1 bias-fold block (h = 1)
#define HSTR   68      // padded stride for h staging in LDS

typedef __attribute__((ext_vector_type(16))) _Float16 v16h;
typedef __attribute__((ext_vector_type(8)))  _Float16 v8h;
typedef __attribute__((ext_vector_type(8)))  float    v8f;

// LDS budget for the edge-GEMM kernel
#define BLDS_HALFS   (KSTEPS * 2 * 32 * 16)            // 66560 halves = 133120 B
#define BLDS_BYTES   (BLDS_HALFS * 2)
#define W1T_BYTES    (HID * EDGE_INF * 4)              // 4096  (transposed [k][c])
#define B1_BYTES     (HID * 4)                         // 256
#define HS_BYTES     (8 * 2 * 16 * HSTR * 2)           // 34816
#define GEMM_LDS     (BLDS_BYTES + W1T_BYTES + B1_BYTES + HS_BYTES)   // 172288

// ---------------------------------------------------------------------------
// K0a: pack W2 (+b2 as 65th block) into the WMMA-B lane-fragment layout, f16.
// B tile for step kk is [K=32 (i), N=16 (o)]:
//   lane l<16  -> col o = l,    rows i = 0..15  (2 per VGPR)
//   lane l>=16 -> col o = l-16, rows i = 16..31
// Bpack layout: [net][kk][ntile][lane][16 halves]
// ---------------------------------------------------------------------------
__global__ void DGAT_packB(const float* __restrict__ sw2, const float* __restrict__ sb2,
                           const float* __restrict__ dw2, const float* __restrict__ db2,
                           _Float16* __restrict__ Bpack) {
    int t = blockIdx.x * blockDim.x + threadIdx.x;
    if (t >= 2 * KSTEPS * 2 * 32) return;
    int lane = t & 31;
    int nt   = (t >> 5) & 1;
    int kk   = (t >> 6) % KSTEPS;
    int net  = t / (KSTEPS * 2 * 32);
    const float* w2 = net ? dw2 : sw2;
    const float* b2 = net ? db2 : sb2;
    int col   = nt * 16 + (lane & 15);
    int kbase = (lane < 16) ? 0 : 16;
    _Float16* dst = Bpack + (size_t)t * 16;
#pragma unroll
    for (int v = 0; v < 16; ++v) {
        int K = kbase + v;  // i index within 32-wide block
        float val = (kk < HID) ? w2[(size_t)kk * (IN_F * HO) + K * HO + col]
                               : b2[K * HO + col];
        dst[v] = (_Float16)val;
    }
}

// ---------------------------------------------------------------------------
// K0b: per-node prep: f16 copy of x (L2-resident gather source),
// x_self = x @ self_weights, self attention logit (leaky-relu'ed),
// seg_max init = self logit (self loop guarantees non-empty segments).
// ---------------------------------------------------------------------------
__global__ void DGAT_prep_nodes(const float* __restrict__ x, const float* __restrict__ sw,
                                const float* __restrict__ att,
                                _Float16* __restrict__ xh, float* __restrict__ xself,
                                float* __restrict__ raws, float* __restrict__ segmax) {
    int n = blockIdx.x * blockDim.x + threadIdx.x;
    if (n >= N_NODES) return;
    float xr[IN_F];
#pragma unroll
    for (int i = 0; i < IN_F; ++i) {
        xr[i] = x[(size_t)n * IN_F + i];
        xh[(size_t)n * IN_F + i] = (_Float16)xr[i];
    }
    float xs[HO];
#pragma unroll 4
    for (int o = 0; o < HO; ++o) {
        float a = 0.f;
#pragma unroll
        for (int i = 0; i < IN_F; ++i) a += xr[i] * sw[i * HO + o];
        xs[o] = a;
        xself[(size_t)n * HO + o] = a;
    }
#pragma unroll
    for (int h = 0; h < HEADS; ++h) {
        float r = 0.f;
#pragma unroll
        for (int f = 0; f < OUTF; ++f)
            r += (att[h * 2 * OUTF + f] + att[h * 2 * OUTF + OUTF + f]) * xs[h * OUTF + f];
        r = (r >= 0.f) ? r : NEG_SLOPE * r;
        raws[n * HEADS + h]   = r;
        segmax[n * HEADS + h] = r;
    }
}

// ---------------------------------------------------------------------------
// K1: the big fused kernel.  For each 32-edge pair-tile and net (blockIdx.y):
//   h[e,0:64] = relu(ef[e] @ W1 + b1)   (VALU, co-executes with WMMA)
//   x_{i|j}[e,:] = Z @ Bnet   where Z[e, kk*32+i] = h[e,kk] * xh[j(e), i]
// A fragment per K-step = cached xg fragment * per-lane scalar h -> 8 pk_mul.
// B fragments resident in LDS (130 KB), filled by GLOBAL_LOAD_ASYNC_TO_LDS,
// one 32B ds read per fragment in the hot loop.
// ---------------------------------------------------------------------------
__global__ void __launch_bounds__(256, 1)
DGAT_edge_gemm(const float* __restrict__ w1s, const float* __restrict__ b1s,
               const float* __restrict__ w1d, const float* __restrict__ b1d,
               const int* __restrict__ ei1, const _Float16* __restrict__ xh,
               const _Float16* __restrict__ Bpack, const float* __restrict__ ef,
               float* __restrict__ xiw, float* __restrict__ xjw) {
    extern __shared__ char smem[];
    _Float16* Blds = (_Float16*)smem;
    float*    W1T  = (float*)(smem + BLDS_BYTES);                 // [k][c] transposed
    float*    B1   = (float*)(smem + BLDS_BYTES + W1T_BYTES);
    _Float16* HS   = (_Float16*)(smem + BLDS_BYTES + W1T_BYTES + B1_BYTES);

    const int net = blockIdx.y;
    const float* w1 = net ? w1d : w1s;
    const float* b1 = net ? b1d : b1s;
    float* out = net ? xjw : xiw;
    const _Float16* Bsrc = Bpack + (size_t)net * BLDS_HALFS;

    // --- async B-stage fill: memory -> LDS directly (no VGPR round trip),
    //     tracked by ASYNCcnt.  Low 32 bits of a generic shared pointer are
    //     the LDS byte offset (AS3 -> flat cast keeps offset in the low dword).
    {
        const uint32_t lds_base = (uint32_t)(uintptr_t)Blds;
        for (int t = threadIdx.x; t < BLDS_BYTES / 16; t += blockDim.x) {
            uint32_t ldsa  = lds_base + (uint32_t)t * 16u;
            uint64_t gaddr = (uint64_t)(uintptr_t)(((const uint4*)Bsrc) + t);
            asm volatile("global_load_async_to_lds_b128 %0, %1, off"
                         :: "v"(ldsa), "v"(gaddr) : "memory");
        }
        asm volatile("s_wait_asynccnt 0" ::: "memory");
    }
    for (int t = threadIdx.x; t < HID * EDGE_INF; t += blockDim.x) {
        int k = t / EDGE_INF, c = t % EDGE_INF;
        W1T[t] = w1[c * HID + k];
    }
    if (threadIdx.x < HID) B1[threadIdx.x] = b1[threadIdx.x];
    __syncthreads();

    const int wave = threadIdx.x >> 5;
    const int lane = threadIdx.x & 31;
    const int m    = lane & 15;       // edge row within 16-tile
    const int half = lane >> 4;
    _Float16* hsw = HS + wave * (2 * 16 * HSTR);

    const int npairs = N_EDGES / 32;  // 6250, exact
    for (int p = blockIdx.x * 8 + wave; p < npairs; p += gridDim.x * 8) {
        const int e0 = p * 32;
        int pn = p + gridDim.x * 8;
        if (pn < npairs) __builtin_prefetch(ef + (size_t)pn * 32 * EDGE_INF, 0, 1);

        // --- gather A-base fragments for the two 16-edge subtiles ---
        int ja = ei1[e0 + m];
        int jb = ei1[e0 + 16 + m];
        const _Float16* ra = xh + (size_t)ja * IN_F + half * 16;
        const _Float16* rb = xh + (size_t)jb * IN_F + half * 16;
        v8h alo = *(const v8h*)ra, ahi = *(const v8h*)(ra + 8);
        v8h blo = *(const v8h*)rb, bhi = *(const v8h*)(rb + 8);
        v16h aBaseA, aBaseB;
#pragma unroll
        for (int i = 0; i < 8; ++i) {
            aBaseA[i] = alo[i]; aBaseA[i + 8] = ahi[i];
            aBaseB[i] = blo[i]; aBaseB[i + 8] = bhi[i];
        }

        // --- h = relu(ef@W1+b1), staged to wave-private LDS as f16 ---
        float efra[EDGE_INF], efrb[EDGE_INF];
        const float* efa = ef + (size_t)(e0 + m) * EDGE_INF;
        const float* efb = ef + (size_t)(e0 + 16 + m) * EDGE_INF;
#pragma unroll
        for (int c = 0; c < EDGE_INF; ++c) { efra[c] = efa[c]; efrb[c] = efb[c]; }
        const int k0 = half * 32;
#pragma unroll 4
        for (int kk = 0; kk < 32; ++kk) {
            int k = k0 + kk;
            float ha = B1[k], hb = ha;
#pragma unroll
            for (int c = 0; c < EDGE_INF; ++c) {
                float w = W1T[k * EDGE_INF + c];
                ha += efra[c] * w;
                hb += efrb[c] * w;
            }
            hsw[(0 * 16 + m) * HSTR + k] = (_Float16)fmaxf(ha, 0.f);
            hsw[(1 * 16 + m) * HSTR + k] = (_Float16)fmaxf(hb, 0.f);
        }
        if (half == 0) {   // bias-fold block: h == 1
            hsw[(0 * 16 + m) * HSTR + 64] = (_Float16)1.0f;
            hsw[(1 * 16 + m) * HSTR + 64] = (_Float16)1.0f;
        }
        __builtin_amdgcn_wave_barrier();
        asm volatile("s_wait_dscnt 0" ::: "memory");   // DS in-order per wave

        // --- Z-GEMM: 65 K-steps x (2 subtiles x 2 N-tiles) WMMAs ---
        v8f acc00 = {}, acc01 = {}, acc10 = {}, acc11 = {};
#pragma unroll 13
        for (int kk = 0; kk < KSTEPS; ++kk) {
            _Float16 hA = hsw[(0 * 16 + m) * HSTR + kk];
            _Float16 hB = hsw[(1 * 16 + m) * HSTR + kk];
            v16h aA = aBaseA * hA;   // 8x v_pk_mul_f16
            v16h aB = aBaseB * hB;
            v16h bf0 = *(const v16h*)&Blds[((kk * 2 + 0) * 32 + lane) * 16];
            v16h bf1 = *(const v16h*)&Blds[((kk * 2 + 1) * 32 + lane) * 16];
            acc00 = __builtin_amdgcn_wmma_f32_16x16x32_f16(false, aA, false, bf0, (short)0, acc00, false, false);
            acc01 = __builtin_amdgcn_wmma_f32_16x16x32_f16(false, aA, false, bf1, (short)0, acc01, false, false);
            acc10 = __builtin_amdgcn_wmma_f32_16x16x32_f16(false, aB, false, bf0, (short)0, acc10, false, false);
            acc11 = __builtin_amdgcn_wmma_f32_16x16x32_f16(false, aB, false, bf1, (short)0, acc11, false, false);
        }

        // --- store C/D: lane L, reg r -> row = r + 8*half, col = (L&15)+16*nt
        float* oA = out + (size_t)e0 * HO;
#pragma unroll
        for (int r = 0; r < 8; ++r) {
            int row = r + half * 8;
            oA[row * HO + m]             = acc00[r];
            oA[row * HO + 16 + m]        = acc01[r];
            oA[(16 + row) * HO + m]      = acc10[r];
            oA[(16 + row) * HO + 16 + m] = acc11[r];
        }
    }
}

// ---------------------------------------------------------------------------
// K2: edge attention logits + segment max (sign-split integer atomics).
// ---------------------------------------------------------------------------
__device__ __forceinline__ void atomicMaxF(float* addr, float v) {
    if (v >= 0.f) atomicMax((int*)addr, __float_as_int(v));
    else          atomicMin((unsigned int*)addr, (unsigned int)__float_as_int(v));
}

__global__ void DGAT_raw_edge(const float* __restrict__ xi, const float* __restrict__ xj,
                              const float* __restrict__ att, const int* __restrict__ ei0,
                              float* __restrict__ rawe, float* __restrict__ segmax) {
    int e = blockIdx.x * blockDim.x + threadIdx.x;
    if (e >= N_EDGES) return;
    int d = ei0[e];
#pragma unroll
    for (int h = 0; h < HEADS; ++h) {
        float r = 0.f;
#pragma unroll
        for (int f = 0; f < OUTF; ++f)
            r += att[h * 2 * OUTF + f]        * xi[(size_t)e * HO + h * OUTF + f]
               + att[h * 2 * OUTF + OUTF + f] * xj[(size_t)e * HO + h * OUTF + f];
        r = (r >= 0.f) ? r : NEG_SLOPE * r;
        rawe[(size_t)e * HEADS + h] = r;
        atomicMaxF(&segmax[(size_t)d * HEADS + h], r);
    }
}

// K3: init denom/out accumulators with the self-loop term.
__global__ void DGAT_self_init(const float* __restrict__ raws, const float* __restrict__ segmax,
                               const float* __restrict__ xself,
                               float* __restrict__ denom, float* __restrict__ outacc) {
    int n = blockIdx.x * blockDim.x + threadIdx.x;
    if (n >= N_NODES) return;
#pragma unroll
    for (int h = 0; h < HEADS; ++h) {
        float es = __expf(raws[n * HEADS + h] - segmax[n * HEADS + h]);
        denom[n * HEADS + h] = es;
#pragma unroll
        for (int f = 0; f < OUTF; ++f)
            outacc[(size_t)n * HO + h * OUTF + f] = es * xself[(size_t)n * HO + h * OUTF + f];
    }
}

// K4: scatter exp-weighted messages.
__global__ void DGAT_scatter(const float* __restrict__ rawe, const float* __restrict__ segmax,
                             const float* __restrict__ xj, const int* __restrict__ ei0,
                             float* __restrict__ denom, float* __restrict__ outacc) {
    int e = blockIdx.x * blockDim.x + threadIdx.x;
    if (e >= N_EDGES) return;
    int d = ei0[e];
#pragma unroll
    for (int h = 0; h < HEADS; ++h) {
        float ex = __expf(rawe[(size_t)e * HEADS + h] - segmax[(size_t)d * HEADS + h]);
        atomicAdd(&denom[(size_t)d * HEADS + h], ex);
#pragma unroll
        for (int f = 0; f < OUTF; ++f)
            atomicAdd(&outacc[(size_t)d * HO + h * OUTF + f],
                      ex * xj[(size_t)e * HO + h * OUTF + f]);
    }
}

// K5: normalize + bias.
__global__ void DGAT_final(const float* __restrict__ outacc, const float* __restrict__ denom,
                           const float* __restrict__ bias, float* __restrict__ out) {
    int n = blockIdx.x * blockDim.x + threadIdx.x;
    if (n >= N_NODES) return;
#pragma unroll
    for (int c = 0; c < HO; ++c)
        out[(size_t)n * HO + c] =
            outacc[(size_t)n * HO + c] / (denom[n * HEADS + (c >> 3)] + 1e-16f) + bias[c];
}

// ---------------------------------------------------------------------------
extern "C" void kernel_launch(void* const* d_in, const int* in_sizes, int n_in,
                              void* d_out, int out_size, void* d_ws, size_t ws_size,
                              hipStream_t stream) {
    const float* x     = (const float*)d_in[0];
    const float* ef    = (const float*)d_in[1];
    const int*   ei    = (const int*)  d_in[2];
    const float* sw1   = (const float*)d_in[3];
    const float* sb1   = (const float*)d_in[4];
    const float* sw2   = (const float*)d_in[5];
    const float* sb2   = (const float*)d_in[6];
    const float* dw1   = (const float*)d_in[7];
    const float* db1   = (const float*)d_in[8];
    const float* dw2   = (const float*)d_in[9];
    const float* db2   = (const float*)d_in[10];
    const float* selfw = (const float*)d_in[11];
    const float* att   = (const float*)d_in[12];
    const float* bias  = (const float*)d_in[13];
    const int* ei0 = ei;
    const int* ei1 = ei + N_EDGES;

    // workspace carve (256B aligned)
    char* ws = (char*)d_ws;
    size_t off = 0;
    auto carve = [&](size_t bytes) { char* p = ws + off; off = (off + bytes + 255) & ~(size_t)255; return p; };
    _Float16* Bpack = (_Float16*)carve((size_t)2 * BLDS_HALFS * 2);
    _Float16* xh    = (_Float16*)carve((size_t)N_NODES * IN_F * 2);
    float* xself    = (float*)carve((size_t)N_NODES * HO * 4);
    float* raws     = (float*)carve((size_t)N_NODES * HEADS * 4);
    float* segmax   = (float*)carve((size_t)N_NODES * HEADS * 4);
    float* denom    = (float*)carve((size_t)N_NODES * HEADS * 4);
    float* outacc   = (float*)carve((size_t)N_NODES * HO * 4);
    float* xi       = (float*)carve((size_t)N_EDGES * HO * 4);
    float* xj       = (float*)carve((size_t)N_EDGES * HO * 4);
    float* rawe     = (float*)carve((size_t)N_EDGES * HEADS * 4);
    (void)ws_size; (void)n_in; (void)in_sizes; (void)out_size;

    const int TPB = 256;
    DGAT_packB<<<(2 * KSTEPS * 2 * 32 + TPB - 1) / TPB, TPB, 0, stream>>>(sw2, sb2, dw2, db2, Bpack);
    DGAT_prep_nodes<<<(N_NODES + TPB - 1) / TPB, TPB, 0, stream>>>(x, selfw, att, xh, xself, raws, segmax);

    (void)hipFuncSetAttribute((const void*)DGAT_edge_gemm,
                              hipFuncAttributeMaxDynamicSharedMemorySize, GEMM_LDS);
    DGAT_edge_gemm<<<dim3(128, 2), TPB, GEMM_LDS, stream>>>(sw1, sb1, dw1, db1, ei1, xh, Bpack, ef, xi, xj);

    DGAT_raw_edge<<<(N_EDGES + TPB - 1) / TPB, TPB, 0, stream>>>(xi, xj, att, ei0, rawe, segmax);
    DGAT_self_init<<<(N_NODES + TPB - 1) / TPB, TPB, 0, stream>>>(raws, segmax, xself, denom, outacc);
    DGAT_scatter<<<(N_EDGES + TPB - 1) / TPB, TPB, 0, stream>>>(rawe, segmax, xj, ei0, denom, outacc);
    DGAT_final<<<(N_NODES + TPB - 1) / TPB, TPB, 0, stream>>>(outacc, denom, bias, (float*)d_out);
}
